// InteractionBlock_73899207295098
// MI455X (gfx1250) — matrix operand order repaired and checked by reading, not compiled, not run on hardware
//
#include <hip/hip_runtime.h>
#include <hip/hip_bf16.h>
#include <math.h>

// ---------------------------------------------------------------------------
// MI455X (gfx1250) implementation.
// bf16 WMMA (V_WMMA_F32_16X16X32_BF16) w/ f32 accumulation for the two
// 352x352 GEMMs; 64-row blocks (4x16 panels) so each B fragment feeds 4 WMMAs;
// async global->LDS staging (ASYNCcnt) when the toolchain exposes it.
// Roofline: ~490 MB HBM -> ~21us floor @ 23.3 TB/s; bf16 WMMA keeps the
// 4.96e10 GEMM FLOPs near that floor (f32 WMMA 16x16x4 would be ~8x slower).
// ---------------------------------------------------------------------------

typedef __attribute__((ext_vector_type(16))) __bf16 v16bf;
typedef __attribute__((ext_vector_type(8)))  __bf16 v8bf;
typedef __attribute__((ext_vector_type(8)))  float  v8f;
typedef int v4i_vs __attribute__((vector_size(16)));   // matches builtin param

#define NODE_DIM 128
#define EQUI_DIM 480
#define CONCAT   352      // 128 + 224
#define KTILES   11       // 352 / 32
#define CTILES   22       // 352 / 16
#define OUT_DIM  608      // 128 + 480
#define EPSV     1e-5f
#define MPANEL   4        // 4 x 16 = 64 rows per block
#define LDS_STRIDE_H  368 // halves per LDS row (352 padded; 736B keeps 16B align)
#define LDS_STRIDE_DW 184

__device__ __forceinline__ __bf16 f2bf(float f) {
  unsigned u = __builtin_bit_cast(unsigned, f);
  unsigned r = u + 0x7FFFu + ((u >> 16) & 1u);   // round-to-nearest-even
  unsigned short h = (unsigned short)(r >> 16);
  return __builtin_bit_cast(__bf16, h);
}

__device__ __forceinline__ float wred(float v) {
#pragma unroll
  for (int o = 16; o > 0; o >>= 1) v += __shfl_xor(v, o, 32);
  return v;
}

// ---- async global->LDS (CDNA5): 16B per call, ASYNCcnt tracked -------------
__device__ __forceinline__ void async_cp16(const char* g, char* l) {
#if __has_builtin(__builtin_amdgcn_global_load_async_to_lds_b128)
  __builtin_amdgcn_global_load_async_to_lds_b128(
      (__attribute__((address_space(1))) v4i_vs*)g,
      (__attribute__((address_space(3))) v4i_vs*)l, 0, 0);
#else
  *(uint4*)l = *(const uint4*)g;
#endif
}

__device__ __forceinline__ void wait_async_lds() {
#if __has_builtin(__builtin_amdgcn_s_wait_asynccnt)
  __builtin_amdgcn_s_wait_asynccnt(0);
#elif __has_builtin(__builtin_amdgcn_global_load_async_to_lds_b128) && defined(__gfx1250__)
  asm volatile("s_wait_asynccnt 0" ::: "memory");
#endif
}

// ---------------------------------------------------------------------------
// Kernel 1: pack f32 weights [352x352] (K-major) into bf16 WMMA-B fragments.
// B packed in contiguous K order (same as A): lanes 0-15 hold K=kt*32+0..15, lanes 16-31
// hold K=kt*32+16..31; column N = ct*16 + (lane&15).
// dst = (((ct*11)+kt)*32 + lane)*16 + h  -> one contiguous 32B load per lane.
// ---------------------------------------------------------------------------
__global__ __launch_bounds__(256)
void pack_w_kernel(const float* __restrict__ W, __bf16* __restrict__ Wp) {
  int idx = blockIdx.x * 256 + threadIdx.x;
  if (idx >= CTILES * KTILES * 32 * 16) return;
  int h    = idx & 15;
  int lane = (idx >> 4) & 31;
  int t    = idx >> 9;            // ct*11 + kt
  int kt   = t % KTILES;
  int ct   = t / KTILES;
  int K    = kt * 32 + ((lane >= 16) ? 16 : 0) + h;
  int Ncol = ct * 16 + (lane & 15);
  Wp[idx] = f2bf(W[(size_t)K * CONCAT + Ncol]);
}

// ---------------------------------------------------------------------------
// Kernel 2: per-row prep, one wave32 per row. Emits bf16 cat row [352] and
// 4 stats/row (mu0, inv0, inv1, inv2) so eb is recomputable in the epilogue.
// ---------------------------------------------------------------------------
__global__ __launch_bounds__(256)
void prep_kernel(const float* __restrict__ ns_in, const float* __restrict__ eq_in,
                 const float* __restrict__ ln_w, const float* __restrict__ ln_b,
                 __bf16* __restrict__ catbf, float* __restrict__ stats, int N) {
  int wave = threadIdx.x >> 5;
  int lane = threadIdx.x & 31;
  int row  = blockIdx.x * 8 + wave;
  if (row >= N) return;

  const float* xs = ns_in + (size_t)row * NODE_DIM;
  const float* xe = eq_in + (size_t)row * EQUI_DIM;
  __bf16* crow = catbf + (size_t)row * CONCAT;

  // ---- LayerNorm over 128 scalars (4 per lane) ----
  float4 s = *(const float4*)(xs + lane * 4);
  float mu  = wred(s.x + s.y + s.z + s.w) * (1.f / 128.f);
  float var = wred(s.x * s.x + s.y * s.y + s.z * s.z + s.w * s.w) * (1.f / 128.f) - mu * mu;
  float rs = rsqrtf(var + EPSV);
  float4 w4 = *(const float4*)(ln_w + lane * 4);
  float4 b4 = *(const float4*)(ln_b + lane * 4);
  crow[lane * 4 + 0] = f2bf((s.x - mu) * rs * w4.x + b4.x);
  crow[lane * 4 + 1] = f2bf((s.y - mu) * rs * w4.y + b4.y);
  crow[lane * 4 + 2] = f2bf((s.z - mu) * rs * w4.z + b4.z);
  crow[lane * 4 + 3] = f2bf((s.w - mu) * rs * w4.w + b4.w);

  // ---- Block 0: 128 muls, l=0 (center over muls, RMS over muls) ----
  float4 e = *(const float4*)(xe + lane * 4);
  float mu0 = wred(e.x + e.y + e.z + e.w) * (1.f / 128.f);
  float c0 = e.x - mu0, c1 = e.y - mu0, c2 = e.z - mu0, c3 = e.w - mu0;
  float n2 = wred(c0 * c0 + c1 * c1 + c2 * c2 + c3 * c3) * (1.f / 128.f);
  float inv0 = rsqrtf(n2 + EPSV);
  float g0 = c0 * inv0, g1 = c1 * inv0, g2 = c2 * inv0, g3 = c3 * inv0;
  crow[128 + lane * 4 + 0] = f2bf(g0 * g0);
  crow[128 + lane * 4 + 1] = f2bf(g1 * g1);
  crow[128 + lane * 4 + 2] = f2bf(g2 * g2);
  crow[128 + lane * 4 + 3] = f2bf(g3 * g3);

  // ---- Block 1: 64 muls x 3 comps (2 muls per lane) ----
  const float* x1 = xe + 128 + lane * 6;
  float a0 = x1[0], a1 = x1[1], a2 = x1[2], a3 = x1[3], a4 = x1[4], a5 = x1[5];
  float q0 = a0 * a0 + a1 * a1 + a2 * a2;
  float q1 = a3 * a3 + a4 * a4 + a5 * a5;
  float n21 = wred(q0 + q1) * (1.f / 64.f);
  float inv1 = rsqrtf(n21 + EPSV);
  float s1 = inv1 * inv1 * 0.57735027f;       // 1/sqrt(3)
  crow[256 + lane * 2 + 0] = f2bf(q0 * s1);
  crow[256 + lane * 2 + 1] = f2bf(q1 * s1);

  // ---- Block 2: 32 muls x 5 comps (1 mul per lane) ----
  const float* x2 = xe + 320 + lane * 5;
  float q2 = 0.f;
#pragma unroll
  for (int j = 0; j < 5; ++j) { float v = x2[j]; q2 += v * v; }
  float n22 = wred(q2) * (1.f / 32.f);
  float inv2 = rsqrtf(n22 + EPSV);
  crow[320 + lane] = f2bf(q2 * inv2 * inv2 * 0.44721360f);  // 1/sqrt(5)

  if (lane == 0) {
    float* st = stats + (size_t)row * 4;
    st[0] = mu0; st[1] = inv0; st[2] = inv1; st[3] = inv2;
  }
}

// ---------------------------------------------------------------------------
// Stage MPANEL x 16 rows (bf16, 352 wide) into LDS via async copies.
// LDS row stride 368 halves (736B): 16B-aligned ds_load_b128, low-conflict.
// ---------------------------------------------------------------------------
__device__ __forceinline__ void stage_panels(unsigned* ldsA,
                                             const __bf16* __restrict__ src,
                                             int rbase, int N) {
  // 44 x 16B chunks per 352-half row
  for (int idx = threadIdx.x; idx < MPANEL * 16 * 44; idx += 256) {
    int r = idx / 44, c = idx % 44;
    int row = rbase + r; if (row >= N) row = N - 1;
    const char* g = (const char*)src + (size_t)row * (CONCAT * 2) + c * 16;
    char* l = (char*)ldsA + (size_t)r * (LDS_STRIDE_DW * 4) + c * 16;
    async_cp16(g, l);
  }
  wait_async_lds();
  __syncthreads();
}

// ---------------------------------------------------------------------------
// WMMA core: one wave computes MPANEL 16x16 f32 tiles for one col-tile.
// A fragment (contiguous K order, matching packed B):
//   lanes 0-15 : row=lane,    K 0-15
//   lanes 16-31: row=lane-16, K 16-31
// Each B fragment (one 32B load/lane) feeds MPANEL WMMAs.
// ---------------------------------------------------------------------------
__device__ __forceinline__ void gemm_tiles(const unsigned* ldsA,
                                           const __bf16* __restrict__ Wp,
                                           int ct, int lane, v8f acc[MPANEL]) {
  const int half = (lane >= 16) ? 16 : 0;
  const int arow = lane & 15;
#pragma unroll
  for (int kt = 0; kt < KTILES; ++kt) {
    v16bf b = *(const v16bf*)(Wp + (size_t)((ct * KTILES + kt) * 32 + lane) * 16);
#pragma unroll
    for (int p = 0; p < MPANEL; ++p) {
      const __bf16* ap = (const __bf16*)ldsA +
                         (size_t)(p * 16 + arow) * LDS_STRIDE_H + kt * 32 + half;
      v8bf alo = *(const v8bf*)(ap);        // ds_load_b128
      v8bf ahi = *(const v8bf*)(ap + 8);    // ds_load_b128
      v16bf a = __builtin_shufflevector(alo, ahi, 0, 1, 2, 3, 4, 5, 6, 7,
                                        8, 9, 10, 11, 12, 13, 14, 15);
      acc[p] = __builtin_amdgcn_wmma_f32_16x16x32_bf16(false, a, false, b,
                                                       (short)0, acc[p],
                                                       false, false);
    }
  }
}

// ---------------------------------------------------------------------------
// Kernel 3: h1 = silu(cat @ W1 + b1), bf16 out. 64 rows per block.
// ---------------------------------------------------------------------------
__global__ __launch_bounds__(256)
void gemm1_kernel(const __bf16* __restrict__ Abf, const __bf16* __restrict__ Wp,
                  const float* __restrict__ bias, __bf16* __restrict__ Hbf, int N) {
  __shared__ __align__(16) unsigned ldsA[MPANEL * 16 * LDS_STRIDE_DW];
  const int rbase = blockIdx.x * (MPANEL * 16);
  stage_panels(ldsA, Abf, rbase, N);
  const int wave = threadIdx.x >> 5, lane = threadIdx.x & 31;
  for (int ct = wave; ct < CTILES; ct += 8) {
    v8f acc[MPANEL];
#pragma unroll
    for (int p = 0; p < MPANEL; ++p)
      acc[p] = (v8f){0.f, 0.f, 0.f, 0.f, 0.f, 0.f, 0.f, 0.f};
    gemm_tiles(ldsA, Wp, ct, lane, acc);
    const int col = ct * 16 + (lane & 15);
    const float bcol = bias[col];
#pragma unroll
    for (int p = 0; p < MPANEL; ++p) {
      const int rb = rbase + p * 16 + ((lane >= 16) ? 8 : 0);
#pragma unroll
      for (int i = 0; i < 8; ++i) {
        int row = rb + i;
        if (row < N) {
          float x = acc[p][i] + bcol;
          float y = x / (1.f + __expf(-x));     // SiLU
          Hbf[(size_t)row * CONCAT + col] = f2bf(y);
        }
      }
    }
  }
}

// ---------------------------------------------------------------------------
// Kernel 4: h = h1 @ W2 + b2, fused gate + residual epilogue -> out[N,608].
// ---------------------------------------------------------------------------
__global__ __launch_bounds__(256)
void gemm2_kernel(const __bf16* __restrict__ Hbf, const __bf16* __restrict__ Wp,
                  const float* __restrict__ bias,
                  const float* __restrict__ ns_in, const float* __restrict__ eq_in,
                  const float* __restrict__ stats, float* __restrict__ out, int N) {
  __shared__ __align__(16) unsigned ldsA[MPANEL * 16 * LDS_STRIDE_DW];
  const int rbase = blockIdx.x * (MPANEL * 16);
  stage_panels(ldsA, Hbf, rbase, N);
  const int wave = threadIdx.x >> 5, lane = threadIdx.x & 31;
  for (int ct = wave; ct < CTILES; ct += 8) {
    v8f acc[MPANEL];
#pragma unroll
    for (int p = 0; p < MPANEL; ++p)
      acc[p] = (v8f){0.f, 0.f, 0.f, 0.f, 0.f, 0.f, 0.f, 0.f};
    gemm_tiles(ldsA, Wp, ct, lane, acc);
    const int col = ct * 16 + (lane & 15);
    const float bcol = bias[col];
#pragma unroll
    for (int p = 0; p < MPANEL; ++p) {
      const int rb = rbase + p * 16 + ((lane >= 16) ? 8 : 0);
#pragma unroll
      for (int i = 0; i < 8; ++i) {
        int row = rb + i;
        if (row >= N) continue;
        float h = acc[p][i] + bcol;
        float* orow = out + (size_t)row * OUT_DIM;
        if (col < NODE_DIM) {
          orow[col] = ns_in[(size_t)row * NODE_DIM + col] + h;   // d_scalar
        } else {
          int g = col - NODE_DIM;                 // gate index 0..223
          const float* xe = eq_in + (size_t)row * EQUI_DIM;
          const float* st = stats + (size_t)row * 4;
          if (g < 128) {                          // l=0: 1 comp per mul
            float x = xe[g];
            float eb = (x - st[0]) * st[1];
            orow[NODE_DIM + g] = x + eb * h;
          } else if (g < 192) {                   // l=1: 3 comps per mul
            int base = 128 + (g - 128) * 3;
            float inv1 = st[2];
#pragma unroll
            for (int j = 0; j < 3; ++j) {
              float x = xe[base + j];
              orow[NODE_DIM + base + j] = x + x * inv1 * h;
            }
          } else {                                // l=2: 5 comps per mul
            int base = 320 + (g - 192) * 5;
            float inv2 = st[3];
#pragma unroll
            for (int j = 0; j < 5; ++j) {
              float x = xe[base + j];
              orow[NODE_DIM + base + j] = x + x * inv2 * h;
            }
          }
        }
      }
    }
  }
}

// ---------------------------------------------------------------------------
extern "C" void kernel_launch(void* const* d_in, const int* in_sizes, int n_in,
                              void* d_out, int out_size, void* d_ws, size_t ws_size,
                              hipStream_t stream) {
  (void)n_in; (void)out_size; (void)ws_size;
  const float* node_scalar = (const float*)d_in[0];
  const float* node_equi   = (const float*)d_in[1];
  const float* ln_w        = (const float*)d_in[2];
  const float* ln_b        = (const float*)d_in[3];
  const float* W1          = (const float*)d_in[4];
  const float* b1          = (const float*)d_in[5];
  const float* W2          = (const float*)d_in[6];
  const float* b2          = (const float*)d_in[7];
  float* out = (float*)d_out;
  const int N = in_sizes[0] / NODE_DIM;

  // Workspace carve-out (~143 MB for N=100k)
  char* ws = (char*)d_ws;
  size_t off = 0;
  auto carve = [&](size_t bytes) -> char* {
    char* p = ws + off;
    off = (off + bytes + 255) & ~(size_t)255;
    return p;
  };
  __bf16* catbf = (__bf16*)carve((size_t)N * CONCAT * sizeof(__bf16));
  __bf16* h1bf  = (__bf16*)carve((size_t)N * CONCAT * sizeof(__bf16));
  float*  stats = (float*)carve((size_t)N * 4 * sizeof(float));
  const size_t wpackN = (size_t)CTILES * KTILES * 32 * 16;
  __bf16* W1p = (__bf16*)carve(wpackN * sizeof(__bf16));
  __bf16* W2p = (__bf16*)carve(wpackN * sizeof(__bf16));

  const int packBlocks = (int)((wpackN + 255) / 256);
  pack_w_kernel<<<packBlocks, 256, 0, stream>>>(W1, W1p);
  pack_w_kernel<<<packBlocks, 256, 0, stream>>>(W2, W2p);

  prep_kernel<<<(N + 7) / 8, 256, 0, stream>>>(node_scalar, node_equi,
                                               ln_w, ln_b, catbf, stats, N);

  const int mblocks = (N + MPANEL * 16 - 1) / (MPANEL * 16);
  gemm1_kernel<<<mblocks, 256, 0, stream>>>(catbf, W1p, b1, h1bf, N);
  gemm2_kernel<<<mblocks, 256, 0, stream>>>(h1bf, W2p, b2, node_scalar,
                                            node_equi, stats, out, N);
}
